// GCN_79542794322477
// MI455X (gfx1250) — matrix operand order repaired and checked by reading, not compiled, not run on hardware
//
#include <hip/hip_runtime.h>
#include <stdint.h>

typedef __attribute__((ext_vector_type(16))) _Float16 v16h;
typedef __attribute__((ext_vector_type(8)))  float    v8f;
typedef __attribute__((ext_vector_type(4)))  float    f32x4;

__device__ __forceinline__ void atomAddF(float* p, float v) {
  unsafeAtomicAdd(p, v);  // lowers to global_atomic_add_f32 (no CAS loop)
}

// ---------- degree accumulation (into normSrc/normDst buffers) ----------
__global__ void gcn_deg(const int* __restrict__ src, const int* __restrict__ dst,
                        float* degOut, float* degIn, int nE) {
  int e = blockIdx.x * blockDim.x + threadIdx.x;
  if (e >= nE) return;
  atomAddF(&degOut[src[e]], 1.0f);
  atomAddF(&degIn[dst[e]], 1.0f);
}

// ---------- deg -> rsqrt(clip(deg,1)) in place; per-graph node counts ----------
__global__ void gcn_norm(float* normSrc, float* normDst, const int* __restrict__ gid,
                         float* cnt, int nN) {
  int n = blockIdx.x * blockDim.x + threadIdx.x;
  if (n >= nN) return;
  float o = normSrc[n]; o = o < 1.f ? 1.f : o;
  float i = normDst[n]; i = i < 1.f ? 1.f : i;
  normSrc[n] = rsqrtf(o);
  normDst[n] = rsqrtf(i);
  atomAddF(&cnt[gid[n]], 1.0f);
}

// ---------- weights -> f16, transposed [n][k]; W2 padded to 16 cols ----------
__global__ void gcn_cvt(const float* __restrict__ W1, const float* __restrict__ W2,
                        _Float16* W1t, _Float16* W2t) {
  int t = blockIdx.x * blockDim.x + threadIdx.x;
  if (t < 64 * 64) {
    int n = t >> 6, k = t & 63;
    W1t[n * 64 + k] = (_Float16)W1[k * 64 + n];
  } else if (t < 64 * 64 + 16 * 64) {
    int u = t - 64 * 64;
    int n = u >> 6, k = u & 63;
    W2t[n * 64 + k] = (n < 4) ? (_Float16)W2[k * 4 + n] : (_Float16)0.f;
  }
}

// ---------- layer-1 edge aggregation: AGG1[dst] += norm_src[src]*h[src], 64 feats ----------
__global__ void gcn_scatter64(const float* __restrict__ h, const int* __restrict__ src,
                              const int* __restrict__ dst, const float* __restrict__ normSrc,
                              float* AGG, int nE) {
  int t = blockIdx.x * blockDim.x + threadIdx.x;
  int e = t >> 4;                 // 16 threads/edge, 4 floats each
  if (e >= nE) return;
  int c = (t & 15) << 2;
  int s = src[e], d = dst[e];
  float ns = normSrc[s];
  const f32x4 v = *(const f32x4*)(h + (size_t)s * 64 + c);
  float* o = AGG + (size_t)d * 64 + c;
  atomAddF(o + 0, v[0] * ns);
  atomAddF(o + 1, v[1] * ns);
  atomAddF(o + 2, v[2] * ns);
  atomAddF(o + 3, v[3] * ns);
}

// ---------- layer-1 GEMM: H1 = relu((normDst .* AGG1) @ W1 + b1), WMMA f16 ----------
// Rows are padded to `pad` (>= nTiles*16/4*16); padding rows of AGG1/normDst are
// zeroed each launch, so no clamps/guards needed -> clean 8-store epilogue.
__global__ void gcn_gemm1(const float* __restrict__ AGG1, const float* __restrict__ normDst,
                          const _Float16* __restrict__ W1t, const float* __restrict__ b1,
                          float* __restrict__ H1, int nTiles) {
  int wave = (blockIdx.x * blockDim.x + threadIdx.x) >> 5;
  int lane = threadIdx.x & 31;
  if (wave >= nTiles) return;     // wave-uniform: EXEC all-ones at WMMA
  int rb = (wave >> 2) << 4;      // row block (16 nodes)
  int nt = (wave & 3) << 4;       // N tile (16 of 64 cols)
  int m = lane & 15, hi = lane >> 4;
  int koff = hi * 16, kb = hi * 16;
  int row = rb + m;
  float nd = normDst[row];
  v8f c = {};
#pragma unroll
  for (int ks = 0; ks < 2; ++ks) {
    int k0 = ks * 32;
    union { f32x4 q[4]; float f[16]; } ax;
    const f32x4* ap = (const f32x4*)(AGG1 + (size_t)row * 64 + k0 + koff);
    ax.q[0] = ap[0]; ax.q[1] = ap[1];   // K = koff .. koff+7
    ax.q[2] = ap[2]; ax.q[3] = ap[3];   // K = koff+8 .. koff+15
    v16h a;
#pragma unroll
    for (int i = 0; i < 16; ++i) a[i] = (_Float16)(ax.f[i] * nd);
    v16h b = *(const v16h*)(W1t + (size_t)(nt + m) * 64 + k0 + kb);
    c = __builtin_amdgcn_wmma_f32_16x16x32_f16(false, a, false, b, (short)0, c, false, false);
  }
  int n = nt + m;
  float bias = b1[n];
  float* outp = H1 + (size_t)(rb + hi * 8) * 64 + n;   // D: lane half picks M base
#pragma unroll
  for (int r = 0; r < 8; ++r) {
    float v = c[r] + bias;
    outp[r * 64] = v > 0.f ? v : 0.f;
  }
}

// ---------- layer-2 transform first: T2 = (normSrc .* H1) @ W2  (nN x 4), WMMA ----------
__global__ void gcn_gemm2(const float* __restrict__ H1, const float* __restrict__ normSrc,
                          const _Float16* __restrict__ W2t, float* __restrict__ T2,
                          int rowTiles) {
  int wave = (blockIdx.x * blockDim.x + threadIdx.x) >> 5;
  int lane = threadIdx.x & 31;
  if (wave >= rowTiles) return;
  int rb = wave << 4;
  int m = lane & 15, hi = lane >> 4;
  int koff = hi * 16, kb = hi * 16;
  int row = rb + m;
  float ns = normSrc[row];
  v8f c = {};
#pragma unroll
  for (int ks = 0; ks < 2; ++ks) {
    int k0 = ks * 32;
    union { f32x4 q[4]; float f[16]; } ax;
    const f32x4* ap = (const f32x4*)(H1 + (size_t)row * 64 + k0 + koff);
    ax.q[0] = ap[0]; ax.q[1] = ap[1];
    ax.q[2] = ap[2]; ax.q[3] = ap[3];
    v16h a;
#pragma unroll
    for (int i = 0; i < 16; ++i) a[i] = (_Float16)(ax.f[i] * ns);
    v16h b = *(const v16h*)(W2t + (size_t)m * 64 + k0 + kb);
    c = __builtin_amdgcn_wmma_f32_16x16x32_f16(false, a, false, b, (short)0, c, false, false);
  }
  if (m < 4) {                    // only first 4 output cols are real
    float* outp = T2 + (size_t)(rb + hi * 8) * 4 + m;
#pragma unroll
    for (int r = 0; r < 8; ++r) outp[r * 4] = c[r];
  }
}

// ---------- layer-2 edge aggregation, only 4 feats/edge (16x less traffic) ----------
__global__ void gcn_scatter4(const float* __restrict__ T2, const int* __restrict__ src,
                             const int* __restrict__ dst, float* AGG2, int nE) {
  int e = blockIdx.x * blockDim.x + threadIdx.x;
  if (e >= nE) return;
  int s = src[e], d = dst[e];
  const f32x4 v = *(const f32x4*)(T2 + (size_t)s * 4);
  float* o = AGG2 + (size_t)d * 4;
  atomAddF(o + 0, v[0]);
  atomAddF(o + 1, v[1]);
  atomAddF(o + 2, v[2]);
  atomAddF(o + 3, v[3]);
}

// ---------- out2 = normDst .* AGG2 + b2, pooled into per-graph sums ----------
__global__ void gcn_pool(const float* __restrict__ AGG2, const float* __restrict__ normDst,
                         const float* __restrict__ b2, const int* __restrict__ gid,
                         float* hg, int nN) {
  int n = blockIdx.x * blockDim.x + threadIdx.x;
  if (n >= nN) return;
  int g = gid[n];
  float nd = normDst[n];
  const f32x4 a = *(const f32x4*)(AGG2 + (size_t)n * 4);
  atomAddF(&hg[g * 4 + 0], nd * a[0] + b2[0]);
  atomAddF(&hg[g * 4 + 1], nd * a[1] + b2[1]);
  atomAddF(&hg[g * 4 + 2], nd * a[2] + b2[2]);
  atomAddF(&hg[g * 4 + 3], nd * a[3] + b2[3]);
}

// ---------- mean + softmax over 16 graphs x 4 classes ----------
__global__ void gcn_softmax(const float* __restrict__ hg, const float* __restrict__ cnt,
                            float* __restrict__ out) {
  int g = threadIdx.x;
  if (g >= 16) return;
  float c = cnt[g]; if (c < 1.f) c = 1.f;
  float v0 = hg[g * 4 + 0] / c, v1 = hg[g * 4 + 1] / c;
  float v2 = hg[g * 4 + 2] / c, v3 = hg[g * 4 + 3] / c;
  float mx = fmaxf(fmaxf(v0, v1), fmaxf(v2, v3));
  float e0 = __expf(v0 - mx), e1 = __expf(v1 - mx);
  float e2 = __expf(v2 - mx), e3 = __expf(v3 - mx);
  float s = e0 + e1 + e2 + e3;
  out[g * 4 + 0] = e0 / s; out[g * 4 + 1] = e1 / s;
  out[g * 4 + 2] = e2 / s; out[g * 4 + 3] = e3 / s;
}

extern "C" void kernel_launch(void* const* d_in, const int* in_sizes, int n_in,
                              void* d_out, int out_size, void* d_ws, size_t ws_size,
                              hipStream_t stream) {
  const float* h  = (const float*)d_in[0];
  const int* src  = (const int*)d_in[1];
  const int* dst  = (const int*)d_in[2];
  const int* gid  = (const int*)d_in[3];
  const float* W1 = (const float*)d_in[4];
  const float* b1 = (const float*)d_in[5];
  const float* W2 = (const float*)d_in[6];
  const float* b2 = (const float*)d_in[7];
  float* out = (float*)d_out;

  const int nE = in_sizes[1];
  const int nN = in_sizes[3];
  const size_t pad = ((size_t)nN + 63) & ~(size_t)63;   // row padding, 256B aligned

  // workspace layout (floats)
  float* ws      = (float*)d_ws;
  float* normSrc = ws;                 // pad
  float* normDst = normSrc + pad;      // pad
  float* AGG1    = normDst + pad;      // pad*64
  float* H1      = AGG1 + pad * 64;    // pad*64
  float* T2      = H1 + pad * 64;      // pad*4
  float* AGG2    = T2 + pad * 4;       // pad*4
  float* hg      = AGG2 + pad * 4;     // 64
  float* cnt     = hg + 64;            // 16
  _Float16* W1t  = (_Float16*)((((uintptr_t)(cnt + 16)) + 31) & ~(uintptr_t)31);
  _Float16* W2t  = W1t + 64 * 64;

  // zero all accumulated regions (incl. padding rows) each launch
  hipMemsetAsync(normSrc, 0, (size_t)(66 * pad) * sizeof(float), stream);      // norms + AGG1
  hipMemsetAsync(AGG2,    0, (size_t)(4 * pad + 80) * sizeof(float), stream);  // AGG2 + hg + cnt

  const int B = 256;
  gcn_deg<<<(nE + B - 1) / B, B, 0, stream>>>(src, dst, normSrc, normDst, nE);
  gcn_norm<<<(nN + B - 1) / B, B, 0, stream>>>(normSrc, normDst, gid, cnt, nN);
  gcn_cvt<<<(64 * 64 + 16 * 64 + B - 1) / B, B, 0, stream>>>(W1, W2, W1t, W2t);

  long long st = (long long)nE * 16;
  gcn_scatter64<<<(int)((st + B - 1) / B), B, 0, stream>>>(h, src, dst, normSrc, AGG1, nE);

  int rowTiles = (nN + 15) >> 4;
  int t1 = rowTiles * 4;                                  // 16x16 tiles, layer 1
  gcn_gemm1<<<((t1 * 32) + B - 1) / B, B, 0, stream>>>(AGG1, normDst, W1t, b1, H1, t1);
  gcn_gemm2<<<((rowTiles * 32) + B - 1) / B, B, 0, stream>>>(H1, normSrc, W2t, T2, rowTiles);

  gcn_scatter4<<<(nE + B - 1) / B, B, 0, stream>>>(T2, src, dst, AGG2, nE);
  gcn_pool<<<(nN + B - 1) / B, B, 0, stream>>>(AGG2, normDst, b2, gid, hg, nN);
  gcn_softmax<<<1, 32, 0, stream>>>(hg, cnt, out);
}
